// fc1_50087908606462
// MI455X (gfx1250) — matrix operand order repaired and checked by reading, not compiled, not run on hardware
//
#include <hip/hip_runtime.h>

typedef __attribute__((ext_vector_type(16))) _Float16 v16h;
typedef __attribute__((ext_vector_type(8)))  _Float16 v8h;
typedef __attribute__((ext_vector_type(4)))  _Float16 v4h;
typedef __attribute__((ext_vector_type(8)))  float    v8f;
typedef int v4i __attribute__((vector_size(16)));

#define BM 128
#define BN 128
#define BK 32
#define LDSA_STRIDE 48   // 32 + 16 halves pad: keeps 16B alignment for ds b128
#define LDSB_STRIDE 48

#if __has_builtin(__builtin_amdgcn_global_load_async_to_lds_b128) && \
    __has_builtin(__builtin_amdgcn_s_wait_asynccnt)
#define HAVE_ASYNC_LDS 1
#else
#define HAVE_ASYNC_LDS 0
#endif

// ---------------- prep kernels ----------------

// x (f32, [M x K]) -> Xh (f16, [M x Kp]), zero-padded K tail
__global__ void cvt_pad_x_kernel(const float* __restrict__ x, _Float16* __restrict__ Xh,
                                 int M, int K, int Kp) {
  int quads_per_row = Kp >> 2;
  int idx = blockIdx.x * blockDim.x + threadIdx.x;
  if (idx >= M * quads_per_row) return;
  int row = idx / quads_per_row;
  int c0  = (idx - row * quads_per_row) << 2;
  v4h h;
  #pragma unroll
  for (int j = 0; j < 4; ++j) {
    int col = c0 + j;
    h[j] = (col < K) ? (_Float16)x[(size_t)row * K + col] : (_Float16)0.0f;
  }
  *(v4h*)&Xh[(size_t)row * Kp + c0] = h;
}

// sign-binarize W ([N x K] f32) into f16 [Np x Kp], zero padding (pad contributes 0)
__global__ void binarize_pad_kernel(const float* __restrict__ W, _Float16* __restrict__ Wb,
                                    int N, int K, int Np, int Kp) {
  int idx = blockIdx.x * blockDim.x + threadIdx.x;
  if (idx >= Np * Kp) return;
  int n = idx / Kp;
  int k = idx - n * Kp;
  float v = 0.0f;
  if (n < N && k < K) v = (W[(size_t)n * K + k] >= 0.0f) ? 1.0f : -1.0f;
  Wb[idx] = (_Float16)v;
}

__global__ void pad_bias_kernel(const float* __restrict__ b, float* __restrict__ bp,
                                int N, int Np) {
  int i = blockIdx.x * blockDim.x + threadIdx.x;
  if (i < Np) bp[i] = (i < N) ? b[i] : 0.0f;
}

// ---------------- WMMA GEMM: C = A @ Bw^T + bias, optional ReLU ----------------
// A:  f16 [M x lda], fully padded (no tails)
// Bw: f16 [Np x Kp] row-major == B-operand columns, K-contiguous, pre-binarized
// C:  [M x ldc] (OutT = _Float16 full-store, or float guarded to Nreal cols)
// Block tile 128x128, 8 waves (4 M x 2 N), wave tile 32x64: 2 A-frags x 4 B-frags,
// 8 wmma per wave per 32-K step, double-buffered LDS, async global->LDS copies.

template <typename OutT, bool RELU>
__global__ void __launch_bounds__(256)
gemm_bin_f16(const _Float16* __restrict__ A, int lda, int Kp,
             const _Float16* __restrict__ Bw,
             const float* __restrict__ bias,
             OutT* __restrict__ C, int ldc, int Nreal) {
  __shared__ __align__(16) _Float16 sA[2][BM * LDSA_STRIDE];
  __shared__ __align__(16) _Float16 sB[2][BN * LDSB_STRIDE];

  const int tid  = threadIdx.x;
  const int lane = tid & 31;
  const int wave = tid >> 5;
  const int wm   = wave & 3;     // wave's M tile (0..3), 32 rows each
  const int wn   = wave >> 2;    // wave's N tile (0..1), 64 cols each
  const int lh   = lane >> 4;    // lane half: selects K-half per ISA layout
  const int lm   = lane & 15;

  const int block_m = blockIdx.y * BM;
  const int block_n = blockIdx.x * BN;

  // staging map: thread covers rows sr and sr+64 of both 128-row tiles,
  // 8-half chunk sc; everything is b128, no guards (all operands padded).
  const int sr = tid >> 2;            // 0..63
  const int sc = (tid & 3) * 8;       // 0,8,16,24
  const _Float16* gA0 = A + (size_t)(block_m + sr) * lda + sc;
  const _Float16* gA1 = A + (size_t)(block_m + sr + 64) * lda + sc;
  const _Float16* gB0 = Bw + (size_t)(block_n + sr) * Kp + sc;
  const _Float16* gB1 = Bw + (size_t)(block_n + sr + 64) * Kp + sc;
  const int lA0 = sr * LDSA_STRIDE + sc;
  const int lA1 = (sr + 64) * LDSA_STRIDE + sc;
  const int lB0 = sr * LDSB_STRIDE + sc;
  const int lB1 = (sr + 64) * LDSB_STRIDE + sc;

  // fragment offsets (ISA 7.12.2 lane layouts)
  const int arow = (wm * 32 + lm) * LDSA_STRIDE + lh * 8;   // K 0..7 / 8..15 per lane half
  const int brow = (wn * 64 + lm) * LDSB_STRIDE + lh * 8;   // K 0..7 / 8..15 per lane half

  v8f acc[2][4];
  #pragma unroll
  for (int m = 0; m < 2; ++m)
    #pragma unroll
    for (int n = 0; n < 4; ++n)
      acc[m][n] = (v8f){};

  int buf = 0;

#if HAVE_ASYNC_LDS
  typedef __attribute__((address_space(1))) v4i* glb_p;
  typedef __attribute__((address_space(3))) v4i* lds_p;
  auto stage = [&](int b, int k0) {
    __builtin_amdgcn_global_load_async_to_lds_b128(
        (glb_p)(gA0 + k0), (lds_p)&sA[b][lA0], 0, 0);
    __builtin_amdgcn_global_load_async_to_lds_b128(
        (glb_p)(gA1 + k0), (lds_p)&sA[b][lA1], 0, 0);
    __builtin_amdgcn_global_load_async_to_lds_b128(
        (glb_p)(gB0 + k0), (lds_p)&sB[b][lB0], 0, 0);
    __builtin_amdgcn_global_load_async_to_lds_b128(
        (glb_p)(gB1 + k0), (lds_p)&sB[b][lB1], 0, 0);
  };
  stage(0, 0);
#else
  v8h pA0 = *(const v8h*)gA0;
  v8h pA1 = *(const v8h*)gA1;
  v8h pB0 = *(const v8h*)gB0;
  v8h pB1 = *(const v8h*)gB1;
#endif

  for (int k0 = 0; k0 < Kp; k0 += BK) {
#if HAVE_ASYNC_LDS
    // wait for the buffer staged last iteration, make visible to all waves,
    // then kick off the next async copy so it overlaps this tile's WMMAs.
    __builtin_amdgcn_s_wait_asynccnt(0);
    __syncthreads();
    if (k0 + BK < Kp) stage(buf ^ 1, k0 + BK);
#else
    // register-pipeline: store current tile, issue next tile's global loads,
    // barrier, compute. Double buffering -> one barrier per iteration.
    *(v8h*)&sA[buf][lA0] = pA0;
    *(v8h*)&sA[buf][lA1] = pA1;
    *(v8h*)&sB[buf][lB0] = pB0;
    *(v8h*)&sB[buf][lB1] = pB1;
    if (k0 + BK < Kp) {
      pA0 = *(const v8h*)(gA0 + k0 + BK);
      pA1 = *(const v8h*)(gA1 + k0 + BK);
      pB0 = *(const v8h*)(gB0 + k0 + BK);
      pB1 = *(const v8h*)(gB1 + k0 + BK);
    }
    __syncthreads();
#endif

    const _Float16* a_base = &sA[buf][0];
    const _Float16* b_base = &sB[buf][0];

    v16h af[2];
    #pragma unroll
    for (int m = 0; m < 2; ++m) {
      v8h lo = *(const v8h*)&a_base[arow + m * 16 * LDSA_STRIDE];
      v8h hi = *(const v8h*)&a_base[arow + m * 16 * LDSA_STRIDE + 16];
      af[m] = __builtin_shufflevector(lo, hi, 0,1,2,3,4,5,6,7,8,9,10,11,12,13,14,15);
    }
    v16h bf[4];
    #pragma unroll
    for (int n = 0; n < 4; ++n) {
      v8h lo = *(const v8h*)&b_base[brow + n * 16 * LDSB_STRIDE];
      v8h hi = *(const v8h*)&b_base[brow + n * 16 * LDSB_STRIDE + 16];
      bf[n] = __builtin_shufflevector(lo, hi, 0,1,2,3,4,5,6,7,8,9,10,11,12,13,14,15);
    }

    #pragma unroll
    for (int m = 0; m < 2; ++m)
      #pragma unroll
      for (int n = 0; n < 4; ++n)
        acc[m][n] = __builtin_amdgcn_wmma_f32_16x16x32_f16(
            false, af[m], false, bf[n], (short)0, acc[m][n], false, false);

    buf ^= 1;
  }

  // ---- epilogue: bias + relu + store (C/D layout: VGPR r -> M = r + lh*8, N = lm) ----
  const int row_base = block_m + wm * 32 + lh * 8;
  const int col_base = block_n + wn * 64 + lm;

  #pragma unroll
  for (int n = 0; n < 4; ++n) {
    const int col = col_base + n * 16;
    const float bv = bias[col];
    #pragma unroll
    for (int m = 0; m < 2; ++m) {
      #pragma unroll
      for (int r = 0; r < 8; ++r) {
        float v = acc[m][n][r] + bv;
        if (RELU) v = (v > 0.0f) ? v : 0.0f;
        int row = row_base + m * 16 + r;
        if constexpr (sizeof(OutT) == 2) {
          C[(size_t)row * ldc + col] = (OutT)v;
        } else {
          if (col < Nreal) C[(size_t)row * ldc + col] = v;
        }
      }
    }
  }
}

// ---------------- host launcher ----------------

extern "C" void kernel_launch(void* const* d_in, const int* in_sizes, int n_in,
                              void* d_out, int out_size, void* d_ws, size_t ws_size,
                              hipStream_t stream) {
  (void)n_in; (void)out_size; (void)ws_size;
  const float* x  = (const float*)d_in[0];
  const float* W1 = (const float*)d_in[1];
  const float* b1 = (const float*)d_in[2];
  const float* W2 = (const float*)d_in[3];
  const float* b2 = (const float*)d_in[4];
  const float* W3 = (const float*)d_in[5];
  const float* b3 = (const float*)d_in[6];
  float* out = (float*)d_out;

  const int M = in_sizes[0] / (28 * 28);           // 65536, multiple of 128
  const int K1 = 784,  K1p = 800;
  const int N1 = 1000, N1p = 1024;
  const int K2p = 1024;
  const int N2 = 500,  N2p = 512;
  const int K3p = 512;
  const int N3 = 10,   N3p = 128;                  // padded to BN

  char* ws = (char*)d_ws;
  size_t off = 0;
  auto wsalloc = [&](size_t bytes) {
    char* p = ws + off;
    off = (off + bytes + 255) & ~(size_t)255;
    return p;
  };
  _Float16* Xh  = (_Float16*)wsalloc((size_t)M * K1p * 2);
  _Float16* H1  = (_Float16*)wsalloc((size_t)M * N1p * 2);
  _Float16* H2  = (_Float16*)wsalloc((size_t)M * N2p * 2);
  _Float16* Wb1 = (_Float16*)wsalloc((size_t)N1p * K1p * 2);
  _Float16* Wb2 = (_Float16*)wsalloc((size_t)N2p * K2p * 2);
  _Float16* Wb3 = (_Float16*)wsalloc((size_t)N3p * K3p * 2);
  float* bp1 = (float*)wsalloc((size_t)N1p * 4);
  float* bp2 = (float*)wsalloc((size_t)N2p * 4);
  float* bp3 = (float*)wsalloc((size_t)N3p * 4);

  // input conversion + weight binarization + padding
  {
    int total = M * (K1p / 4);
    cvt_pad_x_kernel<<<(total + 255) / 256, 256, 0, stream>>>(x, Xh, M, K1, K1p);
  }
  int t1 = N1p * K1p;
  binarize_pad_kernel<<<(t1 + 255) / 256, 256, 0, stream>>>(W1, Wb1, N1, K1, N1p, K1p);
  int t2 = N2p * K2p;
  binarize_pad_kernel<<<(t2 + 255) / 256, 256, 0, stream>>>(W2, Wb2, N2, 1000, N2p, K2p);
  int t3 = N3p * K3p;
  binarize_pad_kernel<<<(t3 + 255) / 256, 256, 0, stream>>>(W3, Wb3, N3, 500, N3p, K3p);
  pad_bias_kernel<<<(N1p + 255) / 256, 256, 0, stream>>>(b1, bp1, N1, N1p);
  pad_bias_kernel<<<1, 256, 0, stream>>>(b2, bp2, N2, N2p);
  pad_bias_kernel<<<1, 256, 0, stream>>>(b3, bp3, N3, N3p);

  // layer 1: f16 Xh -> f16 H1, relu
  gemm_bin_f16<_Float16, true>
      <<<dim3(N1p / BN, M / BM), 256, 0, stream>>>(
          Xh, K1p, K1p, Wb1, bp1, H1, N1p, N1p);
  // layer 2: f16 H1 -> f16 H2, relu (padded H1 cols are exact zeros)
  gemm_bin_f16<_Float16, true>
      <<<dim3(N2p / BN, M / BM), 256, 0, stream>>>(
          H1, N1p, K2p, Wb2, bp2, H2, N2p, N2p);
  // layer 3: f16 H2 -> f32 out, no relu, guarded to 10 cols
  gemm_bin_f16<float, false>
      <<<dim3(N3p / BN, M / BM), 256, 0, stream>>>(
          H2, N2p, K3p, Wb3, bp3, out, N3, N3);
}
